// ContrastiveConceptualAlignmentLoss_72980084293639
// MI455X (gfx1250) — 8-bit matrix operand order repaired and checked by reading, compiled, not run on hardware
//
#include <hip/hip_runtime.h>


typedef int v8i __attribute__((ext_vector_type(8)));

static constexpr int kB    = 512;
static constexpr int kC    = 512;
static constexpr float kInvT = 1.0f / 0.07f;

__global__ __launch_bounds__(256) void pack_kernel(
    const float* __restrict__ concepts,
    unsigned char* __restrict__ cpk,
    float* __restrict__ rsum) {
  const int row  = (int)((blockIdx.x * blockDim.x + threadIdx.x) >> 5);
  const int lane = (int)(threadIdx.x & 31);
  if (row >= kB) return;

  const float4* src = (const float4*)(concepts + (size_t)row * kC + lane * 16);
  int sum = 0;
  unsigned int pk[4];
#pragma unroll
  for (int g = 0; g < 4; ++g) {
    float4 f = src[g];
    unsigned int b0 = (f.x == 1.0f) ? 1u : 0u;
    unsigned int b1 = (f.y == 1.0f) ? 1u : 0u;
    unsigned int b2 = (f.z == 1.0f) ? 1u : 0u;
    unsigned int b3 = (f.w == 1.0f) ? 1u : 0u;
    sum += (int)(b0 + b1 + b2 + b3);
    pk[g] = b0 | (b1 << 8) | (b2 << 16) | (b3 << 24);
  }
  *(uint4*)(cpk + (size_t)row * kC + lane * 16) =
      make_uint4(pk[0], pk[1], pk[2], pk[3]);

#pragma unroll
  for (int m = 16; m >= 1; m >>= 1) sum += __shfl_xor(sum, m, 32);
  if (lane == 0) rsum[row] = (float)sum;
}

__global__ __launch_bounds__(256) void jaccard_gemm_kernel(
    const unsigned char* __restrict__ cpk,
    float* __restrict__ inter) {
  const int wave = (int)((blockIdx.x * blockDim.x + threadIdx.x) >> 5);
  const int lane = (int)(threadIdx.x & 31);
  const int ti   = wave >> 5;
  const int tj   = wave & 31;
  const int half = lane >> 4;
  const int l16  = lane & 15;

  const unsigned char* rowA = cpk + (size_t)(ti * 16 + l16) * kC;
  const unsigned char* rowB = cpk + (size_t)(tj * 16 + l16) * kC;

  v8i acc = {0, 0, 0, 0, 0, 0, 0, 0};
  for (int k = 0; k < kC; k += 64) {
    v8i a, b;
#pragma unroll
    for (int g = 0; g < 2; ++g) {
      uint4 d = *(const uint4*)(rowA + k + g * 32 + half * 16);
      a[4 * g]     = (int)d.x;
      a[4 * g + 1] = (int)d.y;
      a[4 * g + 2] = (int)d.z;
      a[4 * g + 3] = (int)d.w;
    }
#pragma unroll
    for (int g = 0; g < 2; ++g) {
      uint4 d = *(const uint4*)(rowB + k + g * 32 + half * 16);
      b[4 * g]     = (int)d.x;
      b[4 * g + 1] = (int)d.y;
      b[4 * g + 2] = (int)d.z;
      b[4 * g + 3] = (int)d.w;
    }
    acc = __builtin_amdgcn_wmma_i32_16x16x64_iu8(false, a, false, b, acc,
                                                 false, false);
  }

  float* orow = inter + (size_t)(ti * 16 + half * 8) * kB + tj * 16 + l16;
#pragma unroll
  for (int r = 0; r < 8; ++r) orow[(size_t)r * kB] = (float)acc[r];
}

__global__ __launch_bounds__(256) void kl_row_kernel(
    const float* __restrict__ L1,
    const float* __restrict__ L2,
    const float* __restrict__ inter,
    const float* __restrict__ rsum,
    float* __restrict__ contrib) {
  const int i    = (int)((blockIdx.x * blockDim.x + threadIdx.x) >> 5);
  const int lane = (int)(threadIdx.x & 31);
  if (i >= kB) return;

  const float rsi = rsum[i];
  float s[16], x1[16], x2[16];
  float ms = -1e30f, m1 = -1e30f, m2 = -1e30f;
#pragma unroll
  for (int t = 0; t < 16; ++t) {
    const int j    = t * 32 + lane;
    const float iv = inter[(size_t)i * kB + j];
    const float un = rsi + rsum[j] - iv;
    const float sim = (un > 0.0f) ? (iv / un) : 0.0f;
    s[t]  = sim * kInvT;
    x1[t] = L1[(size_t)i * kB + j];
    x2[t] = L2[(size_t)i * kB + j];
    ms = fmaxf(ms, s[t]);
    m1 = fmaxf(m1, x1[t]);
    m2 = fmaxf(m2, x2[t]);
  }
#pragma unroll
  for (int m = 16; m >= 1; m >>= 1) {
    ms = fmaxf(ms, __shfl_xor(ms, m, 32));
    m1 = fmaxf(m1, __shfl_xor(m1, m, 32));
    m2 = fmaxf(m2, __shfl_xor(m2, m, 32));
  }
  float Z = 0.0f, W = 0.0f, Z1 = 0.0f, Z2 = 0.0f;
#pragma unroll
  for (int t = 0; t < 16; ++t) {
    const float w = expf(s[t] - ms);
    Z  += w;
    W  += w * (2.0f * s[t] - x1[t] - x2[t]);
    Z1 += expf(x1[t] - m1);
    Z2 += expf(x2[t] - m2);
  }
#pragma unroll
  for (int m = 16; m >= 1; m >>= 1) {
    Z  += __shfl_xor(Z, m, 32);
    W  += __shfl_xor(W, m, 32);
    Z1 += __shfl_xor(Z1, m, 32);
    Z2 += __shfl_xor(Z2, m, 32);
  }
  if (lane == 0) {
    contrib[i] = W / Z + m1 + logf(Z1) + m2 + logf(Z2)
               - 2.0f * ms - 2.0f * logf(Z);
  }
}

__global__ __launch_bounds__(256) void final_reduce_kernel(
    const float* __restrict__ contrib, float* __restrict__ out) {
  __shared__ float sh[256];
  const int t = (int)threadIdx.x;
  sh[t] = contrib[t] + contrib[t + 256];
  __syncthreads();
#pragma unroll
  for (int off = 128; off >= 1; off >>= 1) {
    if (t < off) sh[t] += sh[t + off];
    __syncthreads();
  }
  if (t == 0) out[0] = sh[0] * (1.0f / (2.0f * (float)kB));
}

extern "C" void kernel_launch(void* const* d_in, const int* in_sizes, int n_in,
                              void* d_out, int out_size, void* d_ws,
                              size_t ws_size, hipStream_t stream) {
  const float* L1       = (const float*)d_in[0];
  const float* L2       = (const float*)d_in[1];
  const float* concepts = (const float*)d_in[2];
  float* out = (float*)d_out;

  float*         inter   = (float*)d_ws;
  unsigned char* cpk     = (unsigned char*)d_ws + (size_t)kB * kB * sizeof(float);
  float*         rsum    = (float*)(cpk + (size_t)kB * kC);
  float*         contrib = rsum + kB;

  pack_kernel<<<kB / 8, 256, 0, stream>>>(concepts, cpk, rsum);
  jaccard_gemm_kernel<<<(kB / 16) * (kB / 16) / 8, 256, 0, stream>>>(cpk, inter);
  kl_row_kernel<<<kB / 8, 256, 0, stream>>>(L1, L2, inter, rsum, contrib);
  final_reduce_kernel<<<1, 256, 0, stream>>>(contrib, out);
}
